// FLASH_85873576116849
// MI455X (gfx1250) — matrix operand order repaired and checked by reading, not compiled, not run on hardware
//
#include <hip/hip_runtime.h>
#include <hip/hip_bf16.h>
#include <math.h>

typedef __attribute__((ext_vector_type(16))) _Float16 v16h;
typedef __attribute__((ext_vector_type(8)))  _Float16 v8h;
typedef __attribute__((ext_vector_type(4)))  _Float16 v4h;
typedef __attribute__((ext_vector_type(8)))  float    v8f;

#define DEV __device__ __forceinline__

static constexpr int DIM   = 1024;
static constexpr int HID   = 2048;
static constexpr int QKD   = 128;
static constexpr int GRP   = 256;
static constexpr int NG    = 16;
static constexpr int BATCH = 4;
static constexpr int SEQ   = 4096;
static constexpr int ROWS  = BATCH * SEQ;   // 16384
static constexpr int NBG   = BATCH * NG;    // 64
static constexpr int LDATT = 264;           // LDS stride (halfs) for attn matrix
static constexpr size_t SMEM_ATTN = (size_t)GRP * LDATT * 2 + GRP * 4;

// ---------------- WMMA helpers (CDNA5 16x16x32 f16 -> f32) ----------------

DEV v8f wmma32(v16h a, v16h b, v8f c) {
  return __builtin_amdgcn_wmma_f32_16x16x32_f16(false, a, false, b,
                                                (short)0, c, false, false);
}

// A fragment: logical A[16][32] row-major at `base` with leading dim `ld` (halfs).
// lane m = lane&15 ; half = lane>>4 : K = half*16 + j  -> contiguous 32B per lane.
DEV v16h frag_a(const _Float16* base, int ld) {
  int lane = threadIdx.x & 31;
  int m = lane & 15, hi = lane >> 4;
  const _Float16* r = base + (size_t)m * ld + (hi << 4);
  v8h lo = *(const v8h*)r;
  v8h hh = *(const v8h*)(r + 8);
  v16h a;
#pragma unroll
  for (int j = 0; j < 8; ++j) { a[j] = lo[j]; a[8 + j] = hh[j]; }
  return a;
}

// B fragment from K-contiguous storage: logical B[k][n] = src[n][k]; src row-major ld.
// lane n = lane&15 ; half = lane>>4 : K = half*16 + j  -> contiguous 32B per lane.
DEV v16h frag_bt(const _Float16* base, int ld) {
  int lane = threadIdx.x & 31;
  int n = lane & 15, hi = lane >> 4;
  const _Float16* r = base + (size_t)n * ld + (hi << 4);
  v8h lo = *(const v8h*)r;
  v8h hh = *(const v8h*)(r + 8);
  v16h b;
#pragma unroll
  for (int j = 0; j < 8; ++j) { b[j] = lo[j]; b[8 + j] = hh[j]; }
  return b;
}

DEV float silu(float x) { return x / (1.0f + __expf(-x)); }

DEV float t5_bias(const float* rel_table, int d) {
  int bucket;
  if (d < 16) bucket = d;
  else {
    int vl = 16 + (int)(logf((float)d * (1.0f / 16.0f)) * (16.0f / 2.0794415416798357f));
    bucket = vl < 31 ? vl : 31;
  }
  return rel_table[bucket] * 11.313708498984761f;  // sqrt(128)
}

// ---------------- Kernel 0: tiled transpose f32[K][N] -> f16[N][K] ----------------

__global__ __launch_bounds__(256)
void flash_wt_transpose(const float* __restrict__ W, _Float16* __restrict__ Wt,
                        int K, int N) {
  __shared__ _Float16 tile[32][33];
  int n0 = blockIdx.x * 32;
  int k0 = blockIdx.y * 32;
  int t = threadIdx.x;
  int c = t & 31, r0 = t >> 5;  // 32 cols x 8 rows, 4 iterations
#pragma unroll
  for (int rr = 0; rr < 32; rr += 8)
    tile[r0 + rr][c] = (_Float16)W[(size_t)(k0 + r0 + rr) * N + n0 + c];
  __syncthreads();
#pragma unroll
  for (int rr = 0; rr < 32; rr += 8)
    Wt[(size_t)(n0 + r0 + rr) * K + k0 + c] = tile[c][r0 + rr];
}

// ---------------- Kernel 1: LayerNorm -> f16 ----------------

__global__ __launch_bounds__(256)
void flash_ln_kernel(const float* __restrict__ x, const float* __restrict__ w,
                     const float* __restrict__ b, _Float16* __restrict__ out) {
  int row = blockIdx.x;
  int t = threadIdx.x;
  const float* xr = x + (size_t)row * DIM;
  float4 v = ((const float4*)xr)[t];
  float s  = v.x + v.y + v.z + v.w;
  float s2 = v.x * v.x + v.y * v.y + v.z * v.z + v.w * v.w;
#pragma unroll
  for (int off = 16; off > 0; off >>= 1) {
    s  += __shfl_down(s,  off);
    s2 += __shfl_down(s2, off);
  }
  __shared__ float ps[8], ps2[8];
  int wv = t >> 5, ln = t & 31;
  if (ln == 0) { ps[wv] = s; ps2[wv] = s2; }
  __syncthreads();
  if (t == 0) {
    float a = 0.f, c = 0.f;
#pragma unroll
    for (int i = 0; i < 8; ++i) { a += ps[i]; c += ps2[i]; }
    ps[0] = a; ps2[0] = c;
  }
  __syncthreads();
  float mean = ps[0] * (1.0f / DIM);
  float var  = ps2[0] * (1.0f / DIM) - mean * mean;
  float rstd = rsqrtf(var + 1e-5f);
  float vals[4] = {v.x, v.y, v.z, v.w};
  v4h o;
#pragma unroll
  for (int j = 0; j < 4; ++j) {
    int c = t * 4 + j;
    o[j] = (_Float16)((vals[j] - mean) * rstd * w[c] + b[c]);
  }
  *(v4h*)(out + (size_t)row * DIM + t * 4) = o;
}

// ---------------- Kernel 2: hv = silu(normed@Wh + bh) -> vT, gate ----------------
// vT layout: [bg][e][n_local]  (K-contiguous for downstream B fragments)

__global__ __launch_bounds__(256)
void flash_hv_gemm(const _Float16* __restrict__ A, const _Float16* __restrict__ Bt,
                   const float* __restrict__ bh, _Float16* __restrict__ vT,
                   _Float16* __restrict__ gatebuf) {
  int n0 = blockIdx.x * 128;
  int m0 = blockIdx.y * 128;
  int t = threadIdx.x, wv = t >> 5, lane = t & 31;
  int fn = lane & 15, mb = (lane >> 4) * 8;
  const _Float16* Ab = A + (size_t)(m0 + wv * 16) * DIM;
  v8f acc[8] = {};
  for (int k0 = 0; k0 < DIM; k0 += 32) {
    __builtin_prefetch((const void*)(Ab + k0 + 64), 0, 1);
    v16h af = frag_a(Ab + k0, DIM);
#pragma unroll
    for (int nt = 0; nt < 8; ++nt) {
      v16h bf = frag_bt(Bt + (size_t)(n0 + nt * 16) * DIM + k0, DIM);
      acc[nt] = wmma32(af, bf, acc[nt]);
    }
  }
#pragma unroll
  for (int nt = 0; nt < 8; ++nt) {
#pragma unroll
    for (int r = 0; r < 8; ++r) {
      int row = m0 + wv * 16 + mb + r;
      int col = n0 + nt * 16 + fn;
      float s = silu(acc[nt][r] + bh[col]);
      if (col < HID) {
        int bg = row >> 8, nl = row & 255;
        vT[((size_t)bg * HID + col) * GRP + nl] = (_Float16)s;
      } else {
        gatebuf[(size_t)row * HID + (col - HID)] = (_Float16)s;
      }
    }
  }
}

// ---------------- Kernel 3: qk = silu(normed@Wqk + bqk) -> 4 heads ----------------
// quadq/linq/quadk: row-major [n][d]; lin_k stored transposed: linkT[bg][d][n_local]

__global__ __launch_bounds__(256)
void flash_qk_gemm(const _Float16* __restrict__ A, const _Float16* __restrict__ Bt,
                   const float* __restrict__ bqk, const float* __restrict__ qkw,
                   const float* __restrict__ qkb, _Float16* __restrict__ quadq,
                   _Float16* __restrict__ linq, _Float16* __restrict__ quadk,
                   _Float16* __restrict__ linkT) {
  int m0 = blockIdx.y * 128;
  int t = threadIdx.x, wv = t >> 5, lane = t & 31;
  int fn = lane & 15, mb = (lane >> 4) * 8;
  const _Float16* Ab = A + (size_t)(m0 + wv * 16) * DIM;
  v8f acc[8] = {};
  for (int k0 = 0; k0 < DIM; k0 += 32) {
    v16h af = frag_a(Ab + k0, DIM);
#pragma unroll
    for (int nt = 0; nt < 8; ++nt) {
      v16h bf = frag_bt(Bt + (size_t)(nt * 16) * DIM + k0, DIM);
      acc[nt] = wmma32(af, bf, acc[nt]);
    }
  }
#pragma unroll
  for (int nt = 0; nt < 8; ++nt) {
#pragma unroll
    for (int r = 0; r < 8; ++r) {
      int row = m0 + wv * 16 + mb + r;
      int col = nt * 16 + fn;
      float s = silu(acc[nt][r] + bqk[col]);
      size_t o = (size_t)row * QKD + col;
      quadq[o] = (_Float16)(s * qkw[0 * QKD + col] + qkb[0 * QKD + col]);
      linq[o]  = (_Float16)(s * qkw[1 * QKD + col] + qkb[1 * QKD + col]);
      quadk[o] = (_Float16)(s * qkw[2 * QKD + col] + qkb[2 * QKD + col]);
      int bg = row >> 8, nl = row & 255;
      linkT[((size_t)bg * QKD + col) * GRP + nl] =
          (_Float16)(s * qkw[3 * QKD + col] + qkb[3 * QKD + col]);
    }
  }
}

// ---------------- Kernel 4: lin_kv[bg] = lin_kT @ v / G -> linkvT[bg][e][d] ----------------

__global__ __launch_bounds__(256)
void flash_linkv_gemm(const _Float16* __restrict__ linkT, const _Float16* __restrict__ vT,
                      _Float16* __restrict__ linkvT) {
  int n0 = blockIdx.x * 128;   // e tile
  int bg = blockIdx.y;
  int t = threadIdx.x, wv = t >> 5, lane = t & 31;
  int fn = lane & 15, mb = (lane >> 4) * 8;
  const _Float16* Ag = linkT + (size_t)bg * QKD * GRP;   // [d][n]
  const _Float16* Bg = vT + (size_t)bg * HID * GRP;      // [e][n]
  v8f acc[8] = {};
  for (int k0 = 0; k0 < GRP; k0 += 32) {
    v16h af = frag_a(Ag + (size_t)(wv * 16) * GRP + k0, GRP);
#pragma unroll
    for (int nt = 0; nt < 8; ++nt) {
      v16h bf = frag_bt(Bg + (size_t)(n0 + nt * 16) * GRP + k0, GRP);
      acc[nt] = wmma32(af, bf, acc[nt]);
    }
  }
#pragma unroll
  for (int nt = 0; nt < 8; ++nt) {
#pragma unroll
    for (int r = 0; r < 8; ++r) {
      int d = wv * 16 + mb + r;
      int e = n0 + nt * 16 + fn;
      linkvT[((size_t)bg * HID + e) * QKD + d] = (_Float16)(acc[nt][r] * (1.0f / GRP));
    }
  }
}

// ---------------- Kernel 5: exclusive cumsum over groups (layout-agnostic) ----------------

__global__ __launch_bounds__(256)
void flash_cumsum(const _Float16* __restrict__ linkvT, _Float16* __restrict__ cumT) {
  size_t idx = (size_t)blockIdx.x * 256 + threadIdx.x;  // b * (HID*QKD) + ed
  size_t per = (size_t)HID * QKD;
  int b = (int)(idx / per);
  size_t ed = idx % per;
  float s = 0.f;
  for (int g = 0; g < NG; ++g) {
    size_t o = ((size_t)(b * NG + g)) * per + ed;
    cumT[o] = (_Float16)s;
    s += (float)linkvT[o];
  }
}

// ---------------- Kernel 6: lin_out = lin_q @ lin_kv_cum -> comb ----------------

__global__ __launch_bounds__(256)
void flash_linout_gemm(const _Float16* __restrict__ linq, const _Float16* __restrict__ cumT,
                       _Float16* __restrict__ comb) {
  int n0 = blockIdx.x * 128;   // e tile
  int m0 = blockIdx.y * 128;   // n-in-group tile (0 or 128)
  int bg = blockIdx.z;
  int t = threadIdx.x, wv = t >> 5, lane = t & 31;
  int fn = lane & 15, mb = (lane >> 4) * 8;
  const _Float16* Ag = linq + (size_t)(bg * GRP + m0 + wv * 16) * QKD;  // [n][d]
  const _Float16* Bg = cumT + (size_t)bg * HID * QKD;                   // [e][d]
  v8f acc[8] = {};
  for (int k0 = 0; k0 < QKD; k0 += 32) {
    v16h af = frag_a(Ag + k0, QKD);
#pragma unroll
    for (int nt = 0; nt < 8; ++nt) {
      v16h bf = frag_bt(Bg + (size_t)(n0 + nt * 16) * QKD + k0, QKD);
      acc[nt] = wmma32(af, bf, acc[nt]);
    }
  }
#pragma unroll
  for (int nt = 0; nt < 8; ++nt) {
#pragma unroll
    for (int r = 0; r < 8; ++r) {
      size_t row = (size_t)bg * GRP + m0 + wv * 16 + mb + r;
      int col = n0 + nt * 16 + fn;
      comb[row * HID + col] = (_Float16)acc[nt][r];
    }
  }
}

// ---------------- Kernel 7: quadratic attention + gated combine ----------------

__global__ __launch_bounds__(256)
void flash_attn_kernel(const _Float16* __restrict__ quadq, const _Float16* __restrict__ quadk,
                       const _Float16* __restrict__ vT, const _Float16* __restrict__ gatebuf,
                       _Float16* __restrict__ comb, const float* __restrict__ rel_table) {
  extern __shared__ char smem[];
  _Float16* attn = (_Float16*)smem;                         // [256][264] f16
  float* biasv = (float*)(smem + (size_t)GRP * LDATT * 2);  // [256] f32
  int echunk = blockIdx.x;   // 0..7 (256-col chunk of HID)
  int bg = blockIdx.y;       // 0..63
  int t = threadIdx.x, wv = t >> 5, lane = t & 31;
  int fn = lane & 15, mb = (lane >> 4) * 8;

  if (t < GRP) biasv[t] = t5_bias(rel_table, t);
  __syncthreads();

  const _Float16* qg = quadq + (size_t)bg * GRP * QKD;
  const _Float16* kg = quadk + (size_t)bg * GRP * QKD;

  // Phase 1: attn = relu(q@kT/G + bias)^2, causal; wave w owns tile-rows 2w,2w+1.
  for (int i = 0; i < 32; ++i) {
    int ti = wv * 2 + (i >> 4);
    int tj = i & 15;
    if (tj > ti) {
#pragma unroll
      for (int r = 0; r < 8; ++r)
        attn[(size_t)(ti * 16 + mb + r) * LDATT + tj * 16 + fn] = (_Float16)0.f;
      continue;
    }
    v8f acc = {};
#pragma unroll
    for (int ks = 0; ks < 4; ++ks) {
      v16h af = frag_a(qg + (size_t)(ti * 16) * QKD + ks * 32, QKD);
      v16h bf = frag_bt(kg + (size_t)(tj * 16) * QKD + ks * 32, QKD);
      acc = wmma32(af, bf, acc);
    }
#pragma unroll
    for (int r = 0; r < 8; ++r) {
      int irow = ti * 16 + mb + r;
      int jcol = tj * 16 + fn;
      int d = irow - jcol;
      float s = 0.f;
      if (d >= 0) {
        s = acc[r] * (1.0f / GRP) + biasv[d];
        s = s > 0.f ? s * s : 0.f;
      }
      attn[(size_t)irow * LDATT + jcol] = (_Float16)s;
    }
  }
  __syncthreads();

  // Phase 2: quad_out = attn @ v (vT is K-contiguous); fuse gate*(quad+lin).
  const _Float16* Bg = vT + (size_t)bg * HID * GRP;  // [e][n]
  for (int i = 0; i < 32; ++i) {
    int ti = wv * 2 + (i >> 4);
    int tj = i & 15;
    int col0 = echunk * 256 + tj * 16;
    v8f acc = {};
#pragma unroll
    for (int ks = 0; ks < 8; ++ks) {
      v16h af = frag_a(attn + (size_t)(ti * 16) * LDATT + ks * 32, LDATT);
      v16h bf = frag_bt(Bg + (size_t)col0 * GRP + ks * 32, GRP);
      acc = wmma32(af, bf, acc);
    }
#pragma unroll
    for (int r = 0; r < 8; ++r) {
      size_t idx = ((size_t)bg * GRP + ti * 16 + mb + r) * HID + col0 + fn;
      float lin = (float)comb[idx];
      float g = (float)gatebuf[idx];
      comb[idx] = (_Float16)(g * (acc[r] + lin));
    }
  }
}

// ---------------- Kernel 8: out = comb @ Wout + bout + x ----------------

__global__ __launch_bounds__(256)
void flash_out_gemm(const _Float16* __restrict__ comb, const _Float16* __restrict__ Bt,
                    const float* __restrict__ bout, const float* __restrict__ x,
                    float* __restrict__ out) {
  int n0 = blockIdx.x * 128;
  int m0 = blockIdx.y * 128;
  int t = threadIdx.x, wv = t >> 5, lane = t & 31;
  int fn = lane & 15, mb = (lane >> 4) * 8;
  const _Float16* Ab = comb + (size_t)(m0 + wv * 16) * HID;
  v8f acc[8] = {};
  for (int k0 = 0; k0 < HID; k0 += 32) {
    __builtin_prefetch((const void*)(Ab + k0 + 64), 0, 1);
    v16h af = frag_a(Ab + k0, HID);
#pragma unroll
    for (int nt = 0; nt < 8; ++nt) {
      v16h bf = frag_bt(Bt + (size_t)(n0 + nt * 16) * HID + k0, HID);
      acc[nt] = wmma32(af, bf, acc[nt]);
    }
  }
#pragma unroll
  for (int nt = 0; nt < 8; ++nt) {
#pragma unroll
    for (int r = 0; r < 8; ++r) {
      int row = m0 + wv * 16 + mb + r;
      int col = n0 + nt * 16 + fn;
      size_t idx = (size_t)row * DIM + col;
      out[idx] = acc[nt][r] + bout[col] + x[idx];
    }
  }
}

// ---------------- Launch ----------------

extern "C" void kernel_launch(void* const* d_in, const int* in_sizes, int n_in,
                              void* d_out, int out_size, void* d_ws, size_t ws_size,
                              hipStream_t stream) {
  (void)in_sizes; (void)n_in; (void)out_size; (void)ws_size;
  const float* x    = (const float*)d_in[0];
  const float* ln_w = (const float*)d_in[1];
  const float* ln_b = (const float*)d_in[2];
  const float* Wh   = (const float*)d_in[3];
  const float* bh   = (const float*)d_in[4];
  const float* Wqk  = (const float*)d_in[5];
  const float* bqk  = (const float*)d_in[6];
  const float* qkw  = (const float*)d_in[7];
  const float* qkb  = (const float*)d_in[8];
  const float* rel  = (const float*)d_in[9];
  const float* Wout = (const float*)d_in[10];
  const float* bout = (const float*)d_in[11];
  float* out = (float*)d_out;

  char* ws = (char*)d_ws;
  size_t off = 0;
  _Float16* normed = (_Float16*)(ws + off); off += (size_t)ROWS * DIM * 2;
  _Float16* vT     = (_Float16*)(ws + off); off += (size_t)ROWS * HID * 2;
  _Float16* gate   = (_Float16*)(ws + off); off += (size_t)ROWS * HID * 2;
  _Float16* quadq  = (_Float16*)(ws + off); off += (size_t)ROWS * QKD * 2;
  _Float16* linq   = (_Float16*)(ws + off); off += (size_t)ROWS * QKD * 2;
  _Float16* quadk  = (_Float16*)(ws + off); off += (size_t)ROWS * QKD * 2;
  _Float16* linkT  = (_Float16*)(ws + off); off += (size_t)ROWS * QKD * 2;
  _Float16* linkvT = (_Float16*)(ws + off); off += (size_t)NBG * QKD * HID * 2;
  _Float16* cumT   = (_Float16*)(ws + off); off += (size_t)NBG * QKD * HID * 2;
  _Float16* comb   = (_Float16*)(ws + off); off += (size_t)ROWS * HID * 2;
  _Float16* Wht    = (_Float16*)(ws + off); off += (size_t)DIM * 2 * HID * 2;
  _Float16* Wqkt   = (_Float16*)(ws + off); off += (size_t)DIM * QKD * 2;
  _Float16* Woutt  = (_Float16*)(ws + off); off += (size_t)HID * DIM * 2;

  // weight pre-transposes (f32 [K][N] -> f16 [N][K])
  flash_wt_transpose<<<dim3(2 * HID / 32, DIM / 32), 256, 0, stream>>>(Wh, Wht, DIM, 2 * HID);
  flash_wt_transpose<<<dim3(QKD / 32, DIM / 32), 256, 0, stream>>>(Wqk, Wqkt, DIM, QKD);
  flash_wt_transpose<<<dim3(DIM / 32, HID / 32), 256, 0, stream>>>(Wout, Woutt, HID, DIM);

  flash_ln_kernel<<<ROWS, 256, 0, stream>>>(x, ln_w, ln_b, normed);
  flash_hv_gemm<<<dim3(2 * HID / 128, ROWS / 128), 256, 0, stream>>>(normed, Wht, bh, vT, gate);
  flash_qk_gemm<<<dim3(1, ROWS / 128), 256, 0, stream>>>(normed, Wqkt, bqk, qkw, qkb,
                                                         quadq, linq, quadk, linkT);
  flash_linkv_gemm<<<dim3(HID / 128, NBG), 256, 0, stream>>>(linkT, vT, linkvT);
  flash_cumsum<<<(BATCH * QKD * HID) / 256, 256, 0, stream>>>(linkvT, cumT);
  flash_linout_gemm<<<dim3(HID / 128, GRP / 128, NBG), 256, 0, stream>>>(linq, cumT, comb);
  flash_attn_kernel<<<dim3(HID / 256, NBG), 256, SMEM_ATTN, stream>>>(quadq, quadk, vT,
                                                                      gate, comb, rel);
  flash_out_gemm<<<dim3(DIM / 128, ROWS / 128), 256, 0, stream>>>(comb, Woutt, bout, x, out);
}
